// MambaBlockV1_29291676958778
// MI455X (gfx1250) — matrix operand order repaired and checked by reading, not compiled, not run on hardware
//
#include <hip/hip_runtime.h>
#include <hip/hip_bf16.h>
#include <math.h>

// ---------------- problem constants ----------------
constexpr int Bc  = 2;
constexpr int Lc  = 4096;
constexpr int Dc  = 256;    // d_model
constexpr int DIc = 512;    // d_inner
constexpr int DSc = 16;     // d_state
constexpr int DTRc = 16;    // dt_rank
constexpr int DCc = 4;      // d_conv
constexpr int XDBL = DTRc + 2 * DSc;   // 48
constexpr int Mrows = Bc * Lc;         // 8192 rows (b*L + t)

typedef _Float16 v16h __attribute__((ext_vector_type(16)));
typedef _Float16 v8h  __attribute__((ext_vector_type(8)));
typedef float    v8f  __attribute__((ext_vector_type(8)));
typedef int      v4i  __attribute__((ext_vector_type(4)));

#if __has_builtin(__builtin_amdgcn_global_load_async_to_lds_b128)
#define USE_ASYNC 1
#else
#define USE_ASYNC 0
#endif

__device__ __forceinline__ float silu_f(float x) {
    return x / (1.0f + __expf(-x));
}
__device__ __forceinline__ float softplus_f(float x) {
    return (x > 20.0f) ? x : log1pf(__expf(x));
}

#if USE_ASYNC
typedef __attribute__((address_space(1))) v4i gv4i;   // global int4
typedef __attribute__((address_space(3))) v4i lv4i;   // LDS int4

__device__ __forceinline__ void async_g2l_b128(const void* g, void* l) {
    __builtin_amdgcn_global_load_async_to_lds_b128((gv4i*)(g), (lv4i*)(l), 0, 0);
}
#endif

template <int N>
__device__ __forceinline__ void wait_async() {
#if __has_builtin(__builtin_amdgcn_s_wait_asynccnt)
    __builtin_amdgcn_s_wait_asynccnt(N);
#else
    asm volatile("s_wait_asynccnt %0" :: "i"(N) : "memory");
#endif
    asm volatile("" ::: "memory");   // keep LDS reads behind the wait
}

// ---------------------------------------------------------------------------
// K0: gather (with optional flip) + LayerNorm -> f16 activations [Mrows, Dc]
// ---------------------------------------------------------------------------
__global__ void gather_ln_f16_kernel(const float* __restrict__ feats,
                                     const int*   __restrict__ perm,
                                     const float* __restrict__ w,
                                     const float* __restrict__ bias,
                                     _Float16* __restrict__ out,
                                     int flip)
{
    const int row = blockIdx.x;            // b*Lc + t
    const int b = row >> 12;
    const int t = row & (Lc - 1);
    const int tp = flip ? (Lc - 1 - t) : t;
    const int pidx = perm[b * Lc + tp];
    const float* src = feats + ((size_t)b * Lc + pidx) * Dc;

    const int c = threadIdx.x;             // 256 threads
    float v = src[c];

    float sum = v, sq = v * v;
    #pragma unroll
    for (int off = 16; off > 0; off >>= 1) {
        sum += __shfl_xor(sum, off, 32);
        sq  += __shfl_xor(sq,  off, 32);
    }
    __shared__ float red[8][2];
    const int wv = threadIdx.x >> 5, ln = threadIdx.x & 31;
    if (ln == 0) { red[wv][0] = sum; red[wv][1] = sq; }
    __syncthreads();
    float ts = 0.f, tq = 0.f;
    #pragma unroll
    for (int i = 0; i < 8; ++i) { ts += red[i][0]; tq += red[i][1]; }
    const float mu  = ts * (1.0f / Dc);
    const float var = tq * (1.0f / Dc) - mu * mu;
    const float r   = rsqrtf(var + 1e-5f);
    out[(size_t)row * Dc + c] = (_Float16)((v - mu) * r * w[c] + bias[c]);
}

// ---------------------------------------------------------------------------
// Pack weight matrix [K,N] f32 (row-major) into per-lane WMMA B fragments.
// ---------------------------------------------------------------------------
__global__ void pack_bfrag_kernel(const float* __restrict__ w,
                                  _Float16* __restrict__ frag,
                                  int K, int N)
{
    const int gid = blockIdx.x * blockDim.x + threadIdx.x;  // K*N/16 threads
    if (gid >= (K * N) / 16) return;
    const int lane = gid & 31;
    const int tile = gid >> 5;
    const int ntiles = N >> 4;
    const int kt = tile / ntiles;
    const int nt = tile - kt * ntiles;
    const int n = nt * 16 + (lane & 15);
    const int kbase = kt * 32 + ((lane >> 4) << 4);
    _Float16* o = frag + (size_t)gid * 16;
    #pragma unroll
    for (int i = 0; i < 16; ++i)
        o[i] = (_Float16)w[(size_t)(kbase + i) * N + n];
}

// ---------------------------------------------------------------------------
// WMMA GEMM: C[M,N] = A[M,K] (f16, row-major) x Bfrag (packed f16), f32 acc.
// block = 128 (4 waves). Each wave -> one 16x64 output strip.
// grid = (M/16, N/256)
//
// Async variant: per-wave double-buffered LDS staging via
// global_load_async_to_lds_b128 (ASYNCcnt), overlapping stage s+1 copies
// with stage s WMMAs. Stage = A tile (1KB) + 4 B fragments (4KB) = 10
// async instructions per wave -> s_wait_asynccnt 10 between stages.
// ---------------------------------------------------------------------------
__global__ void gemm_wmma_f16_kernel(const _Float16* __restrict__ A,
                                     const _Float16* __restrict__ Bf,
                                     float* __restrict__ C,
                                     int M, int N, int K)
{
    const int lane = threadIdx.x & 31;
    const int wave = threadIdx.x >> 5;
    const int mt  = blockIdx.x;
    const int ntg = blockIdx.y * 4 + wave;     // group of 4 n-tiles
    const int n0t = ntg * 4;                   // first n-tile index
    const int m0  = mt * 16;
    const int kg  = (lane >> 4) * 16;          // 0 or 16 (A layout)
    const int ktiles = K >> 5;
    const int ntiles = N >> 4;

    v8f acc[4] = {v8f{}, v8f{}, v8f{}, v8f{}};

#if USE_ASYNC
    // per wave, 2 stages; each stage: A 512 halfs + B 2048 halfs = 5KB
    constexpr int STAGE_H = 512 + 2048;
    __shared__ _Float16 sm[4][2][STAGE_H];

    const int ar = lane & 15;                  // A row this lane copies
    const int ao = (lane >> 4) * 32;           // byte offset within 64B row

    auto issue_stage = [&](int kt, int buf) {
        _Float16* sA = &sm[wave][buf][0];
        _Float16* sB = &sm[wave][buf][512];
        // A tile: 16 rows x 64B, each lane copies 32B (2 x b128)
        const char* ga = (const char*)(A + (size_t)(m0 + ar) * K + kt * 32) + ao;
        char* la = (char*)sA + ar * 64 + ao;
        async_g2l_b128(ga,      la);
        async_g2l_b128(ga + 16, la + 16);
        // B: this wave's 4 tiles are 4KB contiguous in Bf; lane copies 128B
        const char* gb =
            (const char*)(Bf + ((size_t)kt * ntiles + n0t) * 512) + lane * 128;
        char* lb = (char*)sB + lane * 128;
        #pragma unroll
        for (int i = 0; i < 8; ++i)
            async_g2l_b128(gb + i * 16, lb + i * 16);
    };

    issue_stage(0, 0);
    for (int kt = 0; kt < ktiles; ++kt) {
        const int cur = kt & 1;
        if (kt + 1 < ktiles) {
            issue_stage(kt + 1, cur ^ 1);
            wait_async<10>();    // stage `cur` complete; next stage in flight
        } else {
            wait_async<0>();
        }

        const _Float16* sA = &sm[wave][cur][0];
        const _Float16* sB = &sm[wave][cur][512];

        // A fragment from LDS (row-major 16x32 halfs)
        const _Float16* ap = sA + (lane & 15) * 32 + kg;
        v8h alo = *(const v8h*)(ap);
        v8h ahi = *(const v8h*)(ap + 8);
        v16h a;
        #pragma unroll
        for (int i = 0; i < 8; ++i) { a[i] = alo[i]; a[i + 8] = ahi[i]; }

        #pragma unroll
        for (int j = 0; j < 4; ++j) {
            const _Float16* bp = sB + j * 512 + lane * 16;
            v8h blo = *(const v8h*)(bp);
            v8h bhi = *(const v8h*)(bp + 8);
            v16h bfr;
            #pragma unroll
            for (int i = 0; i < 8; ++i) { bfr[i] = blo[i]; bfr[i + 8] = bhi[i]; }
            acc[j] = __builtin_amdgcn_wmma_f32_16x16x32_f16(
                false, a, false, bfr, (short)0, acc[j], false, false);
        }
    }
#else
    const _Float16* arow = A + (size_t)(m0 + (lane & 15)) * K;
    for (int kt = 0; kt < ktiles; ++kt) {
        const _Float16* ap = arow + kt * 32 + kg;
        v8h alo = *(const v8h*)(ap);
        v8h ahi = *(const v8h*)(ap + 8);
        v16h a;
        #pragma unroll
        for (int i = 0; i < 8; ++i) { a[i] = alo[i]; a[i + 8] = ahi[i]; }

        const _Float16* bbase =
            Bf + ((size_t)kt * ntiles + n0t) * 512 + lane * 16;

        if (kt + 1 < ktiles) {
            __builtin_prefetch(arow + (kt + 1) * 32 + kg, 0, 3);
            __builtin_prefetch(Bf + ((size_t)(kt + 1) * ntiles + n0t) * 512 +
                                   lane * 16, 0, 3);
        }

        #pragma unroll
        for (int j = 0; j < 4; ++j) {
            const _Float16* bp = bbase + j * 512;
            v8h blo = *(const v8h*)(bp);
            v8h bhi = *(const v8h*)(bp + 8);
            v16h bfr;
            #pragma unroll
            for (int i = 0; i < 8; ++i) { bfr[i] = blo[i]; bfr[i + 8] = bhi[i]; }
            acc[j] = __builtin_amdgcn_wmma_f32_16x16x32_f16(
                false, a, false, bfr, (short)0, acc[j], false, false);
        }
    }
#endif

    // store D: VGPR i -> row m0 + i + (lane>=16 ? 8 : 0), col (lane&15)
    float* cbase = C + (size_t)(m0 + (lane >> 4) * 8) * N + (lane & 15);
    #pragma unroll
    for (int j = 0; j < 4; ++j) {
        float* cp = cbase + (n0t + j) * 16;
        v8f aj = acc[j];
        #pragma unroll
        for (int i = 0; i < 8; ++i)
            cp[(size_t)i * N] = aj[i];
    }
}

// ---------------------------------------------------------------------------
// K2: causal depthwise conv (DC=4) over t + bias + SiLU. xp = xz[:, 0:DI]
// ---------------------------------------------------------------------------
__global__ void conv_silu_kernel(const float* __restrict__ xz,
                                 const float* __restrict__ cw,
                                 const float* __restrict__ cb,
                                 float* __restrict__ xc)
{
    const int gid = blockIdx.x * blockDim.x + threadIdx.x;  // Mrows*DIc
    const int d = gid & (DIc - 1);
    const int row = gid >> 9;
    const int t = row & (Lc - 1);
    const int b = row >> 12;
    const float* xp = xz + ((size_t)b * Lc) * (2 * DIc) + d;
    float acc = cb[d];
    #pragma unroll
    for (int k = 0; k < DCc; ++k) {
        const int tt = t - (DCc - 1) + k;
        if (tt >= 0)
            acc += cw[d * DCc + k] * xp[(size_t)tt * (2 * DIc)];
    }
    xc[gid] = silu_f(acc);
}

// ---------------------------------------------------------------------------
// K3: x_dbl = xc @ w_x   [Mrows, 48].  One row per block (64 thr, 48 active)
// ---------------------------------------------------------------------------
__global__ void xdbl_kernel(const float* __restrict__ xc,
                            const float* __restrict__ w_x,
                            float* __restrict__ xdbl)
{
    const int row = blockIdx.x;
    const int c = threadIdx.x;
    if (c >= XDBL) return;
    const float* x = xc + (size_t)row * DIc;
    float acc = 0.f;
    for (int k = 0; k < DIc; ++k)
        acc += x[k] * w_x[k * XDBL + c];
    xdbl[(size_t)row * XDBL + c] = acc;
}

// ---------------------------------------------------------------------------
// K4: dt = softplus(dt_r @ w_dt + dt_bias)   [Mrows, DI]
// ---------------------------------------------------------------------------
__global__ void dt_kernel(const float* __restrict__ xdbl,
                          const float* __restrict__ w_dt,
                          const float* __restrict__ dt_bias,
                          float* __restrict__ dt)
{
    const int gid = blockIdx.x * blockDim.x + threadIdx.x;  // Mrows*DIc
    const int d = gid & (DIc - 1);
    const int row = gid >> 9;
    const float* xr = xdbl + (size_t)row * XDBL;
    float acc = dt_bias[d];
    #pragma unroll
    for (int r = 0; r < DTRc; ++r)
        acc += xr[r] * w_dt[r * DIc + d];
    dt[gid] = softplus_f(acc);
}

// ---------------------------------------------------------------------------
// K5: selective scan. One lane per (b, d, s); h kept in a register;
// y = sum_s h*C reduced over the 16-lane state group with shfl_xor (wave32).
// ---------------------------------------------------------------------------
__global__ void scan_kernel(const float* __restrict__ dt,
                            const float* __restrict__ xc,
                            const float* __restrict__ xdbl,
                            const float* __restrict__ a_log,
                            float* __restrict__ ys)
{
    const int g = blockIdx.x * blockDim.x + threadIdx.x;  // Bc*DIc*DSc = 16384
    const int s = g & (DSc - 1);
    const int d = (g >> 4) & (DIc - 1);
    const int b = g >> 13;

    const float A = -__expf(a_log[d * DSc + s]);
    float h = 0.f;

    const float* dtp = dt   + (size_t)b * Lc * DIc + d;
    const float* up  = xc   + (size_t)b * Lc * DIc + d;
    const float* xdp = xdbl + (size_t)b * Lc * XDBL;
    float*       yp  = ys   + (size_t)b * Lc * DIc + d;

    for (int t = 0; t < Lc; ++t) {
        const float dv = dtp[(size_t)t * DIc];
        const float u  = up[(size_t)t * DIc];
        const float Bv = xdp[(size_t)t * XDBL + DTRc + s];
        const float Cv = xdp[(size_t)t * XDBL + DTRc + DSc + s];
        h = __expf(dv * A) * h + (dv * u) * Bv;
        float y = h * Cv;
        y += __shfl_xor(y, 1, 32);
        y += __shfl_xor(y, 2, 32);
        y += __shfl_xor(y, 4, 32);
        y += __shfl_xor(y, 8, 32);
        if (s == 0)
            yp[(size_t)t * DIc] = y;
    }
}

// ---------------------------------------------------------------------------
// K6: ypre = (ys + xc*d_p) * silu(z) -> f16   (z = xz[:, DI:2DI])
// ---------------------------------------------------------------------------
__global__ void ypre_kernel(const float* __restrict__ ys,
                            const float* __restrict__ xc,
                            const float* __restrict__ xz,
                            const float* __restrict__ d_p,
                            _Float16* __restrict__ ypre)
{
    const int gid = blockIdx.x * blockDim.x + threadIdx.x;  // Mrows*DIc
    const int d = gid & (DIc - 1);
    const int row = gid >> 9;
    const float z = xz[(size_t)row * (2 * DIc) + DIc + d];
    const float y = ys[gid] + xc[gid] * d_p[d];
    ypre[gid] = (_Float16)(y * silu_f(z));
}

// ---------------------------------------------------------------------------
// K8: scatter-back (with optional flip) + final LayerNorm -> d_out slice
// ---------------------------------------------------------------------------
__global__ void scatter_ln_kernel(const float* __restrict__ outm,
                                  const int*   __restrict__ perm,
                                  const float* __restrict__ w,
                                  const float* __restrict__ bias,
                                  float* __restrict__ dst,
                                  int flip)
{
    const int row = blockIdx.x;            // b*Lc + t
    const int b = row >> 12;
    const int t = row & (Lc - 1);
    const int tp = flip ? (Lc - 1 - t) : t;
    const int pidx = perm[b * Lc + tp];

    const int c = threadIdx.x;             // 256
    const float v = outm[(size_t)row * Dc + c];

    float sum = v, sq = v * v;
    #pragma unroll
    for (int off = 16; off > 0; off >>= 1) {
        sum += __shfl_xor(sum, off, 32);
        sq  += __shfl_xor(sq,  off, 32);
    }
    __shared__ float red[8][2];
    const int wv = threadIdx.x >> 5, ln = threadIdx.x & 31;
    if (ln == 0) { red[wv][0] = sum; red[wv][1] = sq; }
    __syncthreads();
    float ts = 0.f, tq = 0.f;
    #pragma unroll
    for (int i = 0; i < 8; ++i) { ts += red[i][0]; tq += red[i][1]; }
    const float mu  = ts * (1.0f / Dc);
    const float var = tq * (1.0f / Dc) - mu * mu;
    const float r   = rsqrtf(var + 1e-5f);
    dst[((size_t)b * Lc + pidx) * Dc + c] = (v - mu) * r * w[c] + bias[c];
}

// ---------------------------------------------------------------------------
// host-side launch
// ---------------------------------------------------------------------------
struct BranchParams {
    const float* feats;  const int* perm;  int flip;
    const float* ln_w;   const float* ln_b;
    const float* w_in;   const float* conv_w; const float* conv_b;
    const float* w_x;    const float* w_dt;   const float* dt_bias;
    const float* a_log;  const float* d_p;    const float* w_out;
    const float* fn_w;   const float* fn_b;   // final LN
    float* dst;
};

static void run_branch(const BranchParams& P, char* ws, hipStream_t stream)
{
    size_t off = 0;
    auto carve = [&](size_t bytes) { void* p = ws + off; off += (bytes + 255) & ~(size_t)255; return p; };
    _Float16* xln      = (_Float16*)carve((size_t)Mrows * Dc * 2);
    _Float16* winfrag  = (_Float16*)carve((size_t)Dc * (2 * DIc) * 2);
    _Float16* woutfrag = (_Float16*)carve((size_t)DIc * Dc * 2);
    float*    xz       = (float*)   carve((size_t)Mrows * 2 * DIc * 4);
    float*    xc       = (float*)   carve((size_t)Mrows * DIc * 4);
    float*    xdbl     = (float*)   carve((size_t)Mrows * XDBL * 4);
    float*    dtb      = (float*)   carve((size_t)Mrows * DIc * 4);
    float*    ysb      = (float*)   carve((size_t)Mrows * DIc * 4);
    _Float16* ypre     = (_Float16*)carve((size_t)Mrows * DIc * 2);
    float*    outm     = (float*)   carve((size_t)Mrows * Dc * 4);

    {
        int n1 = (Dc * 2 * DIc) / 16;   // 16384
        pack_bfrag_kernel<<<(n1 + 255) / 256, 256, 0, stream>>>(P.w_in, winfrag, Dc, 2 * DIc);
        int n2 = (DIc * Dc) / 16;       // 8192
        pack_bfrag_kernel<<<(n2 + 255) / 256, 256, 0, stream>>>(P.w_out, woutfrag, DIc, Dc);
    }

    gather_ln_f16_kernel<<<Mrows, Dc, 0, stream>>>(P.feats, P.perm, P.ln_w, P.ln_b, xln, P.flip);

    {
        dim3 grid(Mrows / 16, (2 * DIc) / 256);
        gemm_wmma_f16_kernel<<<grid, 128, 0, stream>>>(xln, winfrag, xz, Mrows, 2 * DIc, Dc);
    }

    conv_silu_kernel<<<(Mrows * DIc) / 256, 256, 0, stream>>>(xz, P.conv_w, P.conv_b, xc);

    xdbl_kernel<<<Mrows, 64, 0, stream>>>(xc, P.w_x, xdbl);
    dt_kernel<<<(Mrows * DIc) / 256, 256, 0, stream>>>(xdbl, P.w_dt, P.dt_bias, dtb);

    scan_kernel<<<(Bc * DIc * DSc) / 256, 256, 0, stream>>>(dtb, xc, xdbl, P.a_log, ysb);

    ypre_kernel<<<(Mrows * DIc) / 256, 256, 0, stream>>>(ysb, xc, xz, P.d_p, ypre);

    {
        dim3 grid(Mrows / 16, Dc / 256);
        gemm_wmma_f16_kernel<<<grid, 128, 0, stream>>>(ypre, woutfrag, outm, Mrows, Dc, DIc);
    }

    scatter_ln_kernel<<<Mrows, Dc, 0, stream>>>(outm, P.perm, P.fn_w, P.fn_b, P.dst, P.flip);
}

extern "C" void kernel_launch(void* const* d_in, const int* in_sizes, int n_in,
                              void* d_out, int out_size, void* d_ws, size_t ws_size,
                              hipStream_t stream)
{
    (void)in_sizes; (void)n_in; (void)out_size; (void)ws_size;
    const float* feats_s1 = (const float*)d_in[0];
    const float* feats_s2 = (const float*)d_in[1];
    const int*   perm_s1  = (const int*)d_in[2];
    const int*   perm_s2  = (const int*)d_in[3];

    auto F = [&](int i) { return (const float*)d_in[i]; };

    float* out = (float*)d_out;
    char* ws = (char*)d_ws;

    // ---- s2 branch: gather(perm_s2) -> mamba layer 1 -> scatter -> LN(norm) ----
    BranchParams p2;
    p2.feats = feats_s2; p2.perm = perm_s2; p2.flip = 0;
    p2.ln_w = F(4);  p2.ln_b = F(5);  p2.w_in = F(6);  p2.conv_w = F(7);
    p2.conv_b = F(8); p2.w_x = F(9); p2.w_dt = F(10); p2.dt_bias = F(11);
    p2.a_log = F(12); p2.d_p = F(13); p2.w_out = F(14);
    p2.fn_w = F(26); p2.fn_b = F(27);
    p2.dst = out;
    run_branch(p2, ws, stream);

    // ---- s1 branch: gather(perm_s1)+flip -> mamba layer 2 -> flip+scatter -> LN(normb) ----
    BranchParams p1;
    p1.feats = feats_s1; p1.perm = perm_s1; p1.flip = 1;
    p1.ln_w = F(15); p1.ln_b = F(16); p1.w_in = F(17); p1.conv_w = F(18);
    p1.conv_b = F(19); p1.w_x = F(20); p1.w_dt = F(21); p1.dt_bias = F(22);
    p1.a_log = F(23); p1.d_p = F(24); p1.w_out = F(25);
    p1.fn_w = F(28); p1.fn_b = F(29);
    p1.dst = out + (size_t)Bc * Lc * Dc;
    run_branch(p1, ws, stream);
}
